// GNN_53979148976171
// MI455X (gfx1250) — matrix operand order repaired and checked by reading, not compiled, not run on hardware
//
#include <hip/hip_runtime.h>
#include <hip/hip_bf16.h>

typedef __attribute__((ext_vector_type(16))) __bf16 v16bf;
typedef __attribute__((ext_vector_type(8)))  float  v8f;

#define EMB  128
#define NCLS 7
#define BN_EPS 1e-5f

static inline int cdiv(long long a, long long b) { return (int)((a + b - 1) / b); }

// ---------------------------------------------------------------------------
// small utility kernels
// ---------------------------------------------------------------------------
__global__ void k_fill(float* __restrict__ p, float v, long long n) {
  long long t = (long long)blockIdx.x * blockDim.x + threadIdx.x;
  if (t < n) p[t] = v;
}

__device__ inline void atomicMaxF(float* addr, float val) {
  int* ai = (int*)addr;
  int old = __float_as_int(*addr);
  while (__int_as_float(old) < val) {
    int assumed = old;
    old = atomicCAS(ai, assumed, __float_as_int(val));
    if (old == assumed) break;
  }
}

// deg[dst] += |ea|   (deg pre-filled with 1.0 = self-loop weight)
__global__ void k_deg(const int* __restrict__ dst, const float* __restrict__ ea,
                      float* __restrict__ deg, int E) {
  int t = blockIdx.x * blockDim.x + threadIdx.x;
  if (t < E) atomicAdd(deg + dst[t], fabsf(ea[t]));
}

// in-place deg -> D^{-1/2}
__global__ void k_dinv(float* __restrict__ deg, int n) {
  int t = blockIdx.x * blockDim.x + threadIdx.x;
  if (t < n) {
    float d = deg[t];
    deg[t] = (d > 0.f) ? rsqrtf(fmaxf(d, 1e-12f)) : 0.f;
  }
}

// per-edge symmetric normalization (layer-invariant, computed once)
__global__ void k_enorm(const int* __restrict__ src, const int* __restrict__ dst,
                        const float* __restrict__ ea, const float* __restrict__ dinv,
                        float* __restrict__ enorm, int E) {
  int t = blockIdx.x * blockDim.x + threadIdx.x;
  if (t < E) enorm[t] = dinv[src[t]] * fabsf(ea[t]) * dinv[dst[t]];
}

__global__ void k_counts(const int* __restrict__ batch, float* __restrict__ cnts, int n) {
  int t = blockIdx.x * blockDim.x + threadIdx.x;
  if (t < n) atomicAdd(cnts + batch[t], 1.0f);
}

// layer-0 projection: hW[n,f] = x[n,0]*W0[0,f] + x[n,1]*W0[1,f]
__global__ void k_in_proj(const float* __restrict__ x, const float* __restrict__ W0,
                          float* __restrict__ hW, int nrows) {
  long long t = (long long)blockIdx.x * blockDim.x + threadIdx.x;
  if (t >= (long long)nrows * EMB) return;
  int n = (int)(t >> 7), f = (int)(t & 127);
  hW[t] = x[2 * n] * W0[f] + x[2 * n + 1] * W0[EMB + f];
}

// ---------------------------------------------------------------------------
// pack W[128,128] f32 into bf16 WMMA B-fragment order:
// Bp[((col_tile*4 + kk_idx)*32 + lane)*16 + idx], idx = (vgpr_pair, lo/hi)
// B 32x16 layout: idx s<8 -> K=8*half+s, s>=8 -> K=16+8*half+(s-8)
//                (half = lane>>4); N = lane&15.
// Each lane's 16 bf16 fragment becomes one contiguous 32-byte chunk.
// ---------------------------------------------------------------------------
__global__ void k_pack_w(const float* __restrict__ W, __bf16* __restrict__ Bp) {
  int t = blockIdx.x * blockDim.x + threadIdx.x;
  if (t >= EMB * EMB) return;
  int idx   = t & 15;
  int lane  = (t >> 4) & 31;
  int kkidx = (t >> 9) & 3;
  int ct    = t >> 11;          // column tile 0..7
  int half  = lane >> 4;
  int v     = idx >> 1, lo = idx & 1;
  int kb    = kkidx * 32 + (half << 3) + (idx & 7) + ((idx >> 3) << 4); (void)v; (void)lo;
  int col   = ct * 16 + (lane & 15);
  Bp[t] = (__bf16)W[kb * EMB + col];
}

// ---------------------------------------------------------------------------
// WMMA GEMM: C[N,128] = A[N,128] (bf16) @ B[128,128] (packed bf16), fp32 acc.
// Block = 8 waves handles 32 rows: A 32x128 tile staged in LDS (8KB); each
// wave owns one 16-col output tile and two 16-row tiles (dual accumulators,
// B fragment reused by both WMMAs). Uniform-guard store epilogue.
// ---------------------------------------------------------------------------
__global__ __launch_bounds__(256)
void k_gemm_wmma(const __bf16* __restrict__ A, const __bf16* __restrict__ Bp,
                 float* __restrict__ C, int nrows) {
  __shared__ __align__(16) __bf16 As[32 * EMB];
  const int tid  = threadIdx.x;
  const int wave = tid >> 5;
  const int lane = tid & 31;
  const int row0 = blockIdx.x * 32;

  // cooperative A tile load: 256 threads x 32 bytes = 32x128 bf16
  {
    int r = tid >> 3;                 // 0..31
    int e = (tid & 7) * 16;           // element 0..112 step 16
    int ra = row0 + r;
    if (ra >= nrows) ra = nrows - 1;  // clamp, keep EXEC full
    *(uint4*)&As[r * EMB + e]     = *(const uint4*)&A[(long long)ra * EMB + e];
    *(uint4*)&As[r * EMB + e + 8] = *(const uint4*)&A[(long long)ra * EMB + e + 8];
  }
  __syncthreads();

  const int m    = lane & 15;
  const int half = lane >> 4;
  v8f acc0 = {}, acc1 = {};
#pragma unroll
  for (int q = 0; q < 4; ++q) {
    const int kk = q * 32;
    // A 16x32 bf16 layout: lane's fragment = two contiguous 16B chunks
    union { uint4 u[2]; v16bf v; } a0, a1;
    a0.u[0] = *(const uint4*)&As[m * EMB + kk + half * 8];
    a0.u[1] = *(const uint4*)&As[m * EMB + kk + 16 + half * 8];
    a1.u[0] = *(const uint4*)&As[(m + 16) * EMB + kk + half * 8];
    a1.u[1] = *(const uint4*)&As[(m + 16) * EMB + kk + 16 + half * 8];
    // packed B fragment: one contiguous 32B chunk per lane, reused twice
    v16bf b = *(const v16bf*)&Bp[(((wave << 2) | q) * 32 + lane) << 4];
    acc0 = __builtin_amdgcn_wmma_f32_16x16x32_bf16(
        false, a0.v, false, b, (short)0, acc0, false, false);
    acc1 = __builtin_amdgcn_wmma_f32_16x16x32_bf16(
        false, a1.v, false, b, (short)0, acc1, false, false);
  }
  // C 16x16 f32 layout: VGPR r: lanes 0-15 -> M=r, lanes 16-31 -> M=8+r
  const int col = (wave << 4) + m;
  const int rbase = row0 + half * 8;
  if (row0 + 32 <= nrows) {           // uniform fast path: branchless stores
#pragma unroll
    for (int r = 0; r < 8; ++r) {
      C[(long long)(rbase + r) * EMB + col]      = acc0[r];
      C[(long long)(rbase + 16 + r) * EMB + col] = acc1[r];
    }
  } else {                            // rare tail block
#pragma unroll
    for (int r = 0; r < 8; ++r) {
      if (rbase + r < nrows)      C[(long long)(rbase + r) * EMB + col]      = acc0[r];
      if (rbase + 16 + r < nrows) C[(long long)(rbase + 16 + r) * EMB + col] = acc1[r];
    }
  }
}

// ---------------------------------------------------------------------------
// edge scatter: out[dst] += hW[src] * enorm[e]  (f32 atomics)
// one thread handles 4 features of one edge (32 threads/edge)
// ---------------------------------------------------------------------------
__global__ void k_scatter(const float* __restrict__ hW, const int* __restrict__ src,
                          const int* __restrict__ dst, const float* __restrict__ enorm,
                          float* __restrict__ out, long long total) {
  long long t = (long long)blockIdx.x * blockDim.x + threadIdx.x;
  if (t >= total) return;
  int e  = (int)(t >> 5);
  int fo = ((int)t & 31) << 2;
  int s = src[e], d = dst[e];
  float nrm = enorm[e];
  const float4 v = *(const float4*)(hW + (long long)s * EMB + fo);
  float* o = out + (long long)d * EMB + fo;
  atomicAdd(o + 0, v.x * nrm);
  atomicAdd(o + 1, v.y * nrm);
  atomicAdd(o + 2, v.z * nrm);
  atomicAdd(o + 3, v.w * nrm);
}

// z = relu(edge_sums + hW*dinv^2 (self loop) + bias); accumulate per-feature stats
__global__ __launch_bounds__(EMB)
void k_bias_relu_stats(float* __restrict__ z, const float* __restrict__ hW,
                       const float* __restrict__ dinv, const float* __restrict__ bias,
                       float* __restrict__ s1, float* __restrict__ s2, int nrows) {
  int f = threadIdx.x;
  float b = bias[f];
  float a1 = 0.f, a2 = 0.f;
  const long long step = (long long)gridDim.x * EMB;
  for (int n = blockIdx.x; n < nrows; n += gridDim.x) {
    float sn = dinv[n]; sn *= sn;
    long long idx = (long long)n * EMB + f;
    // prefetch next strided row (speculative global_prefetch_b8)
    __builtin_prefetch(&z[idx + step], 0, 0);
    __builtin_prefetch(&hW[idx + step], 0, 0);
    float v = z[idx] + hW[idx] * sn + b;
    v = fmaxf(v, 0.f);
    z[idx] = v;
    a1 += v; a2 += v * v;
  }
  atomicAdd(s1 + f, a1);
  atomicAdd(s2 + f, a2);
}

// fold gamma/beta into affine scale/shift (biased variance, as reference)
__global__ void k_bn_finalize(const float* __restrict__ s1, const float* __restrict__ s2,
                              const float* __restrict__ g, const float* __restrict__ beta,
                              float* __restrict__ scale, float* __restrict__ shift,
                              int nrows, int F) {
  int f = blockIdx.x * blockDim.x + threadIdx.x;
  if (f >= F) return;
  float m   = s1[f] / (float)nrows;
  float var = s2[f] / (float)nrows - m * m;
  float sc  = rsqrtf(var + BN_EPS) * g[f];
  scale[f] = sc;
  shift[f] = beta[f] - m * sc;
}

// h_bf = bf16(z*scale+shift); optional pooling atomics (segment max / sum)
__global__ void k_bn_apply_pool(const float* __restrict__ z, const float* __restrict__ scale,
                                const float* __restrict__ shift, __bf16* __restrict__ h_bf,
                                const int* __restrict__ batch, float* __restrict__ gmp,
                                float* __restrict__ gap, int nrows, int do_pool) {
  long long t = (long long)blockIdx.x * blockDim.x + threadIdx.x;
  if (t >= (long long)nrows * EMB) return;
  int n = (int)(t >> 7), f = (int)(t & 127);
  float v = z[t] * scale[f] + shift[f];
  h_bf[t] = (__bf16)v;
  if (do_pool) {
    int b = batch[n];
    atomicAdd(gap + (long long)b * EMB + f, v);
    atomicMaxF(gmp + (long long)b * EMB + f, v);
  }
}

// rep[g, 0:128] += gmp ; rep[g, 128:256] += gap / max(count,1)
__global__ void k_pool_combine(float* __restrict__ rep, const float* __restrict__ gmp,
                               const float* __restrict__ gap, const float* __restrict__ cnts,
                               int G) {
  int t = blockIdx.x * blockDim.x + threadIdx.x;
  if (t >= G * EMB) return;
  int g = t >> 7, f = t & 127;
  float c = fmaxf(cnts[g], 1.0f);
  rep[(long long)g * 2 * EMB + f]       += gmp[t];
  rep[(long long)g * 2 * EMB + EMB + f] += gap[t] / c;
}

// ---------------------------------------------------------------------------
// MLP head
// ---------------------------------------------------------------------------
__global__ void k_fc_relu(const float* __restrict__ in, const float* __restrict__ W,
                          const float* __restrict__ b, float* __restrict__ out,
                          int rows, int I, int O) {
  int t = blockIdx.x * blockDim.x + threadIdx.x;
  if (t >= rows * O) return;
  int g = t / O, j = t % O;
  float acc = b[j];
  for (int k = 0; k < I; ++k) acc += in[(long long)g * I + k] * W[(long long)k * O + j];
  out[t] = fmaxf(acc, 0.f);
}

__global__ void k_colstats(const float* __restrict__ z, float* __restrict__ s1,
                           float* __restrict__ s2, int rows, int F) {
  int f = threadIdx.x;
  if (f >= F) return;
  float a1 = 0.f, a2 = 0.f;
  for (int n = blockIdx.x; n < rows; n += gridDim.x) {
    float v = z[(long long)n * F + f];
    a1 += v; a2 += v * v;
  }
  atomicAdd(s1 + f, a1);
  atomicAdd(s2 + f, a2);
}

__global__ void k_bn_apply(float* __restrict__ z, const float* __restrict__ scale,
                           const float* __restrict__ shift, long long n, int F) {
  long long t = (long long)blockIdx.x * blockDim.x + threadIdx.x;
  if (t >= n) return;
  int f = (int)(t % F);
  z[t] = z[t] * scale[f] + shift[f];
}

__global__ void k_head_out(const float* __restrict__ in, const float* __restrict__ W,
                           const float* __restrict__ b, float* __restrict__ out,
                           int rows, int I) {
  int g = blockIdx.x * blockDim.x + threadIdx.x;
  if (g >= rows) return;
  float logits[NCLS];
  float mx = -3.0e38f;
  for (int j = 0; j < NCLS; ++j) {
    float acc = b[j];
    for (int k = 0; k < I; ++k) acc += in[(long long)g * I + k] * W[k * NCLS + j];
    logits[j] = acc;
    mx = fmaxf(mx, acc);
  }
  float s = 0.f;
  for (int j = 0; j < NCLS; ++j) s += expf(logits[j] - mx);
  float lse = logf(s);
  for (int j = 0; j < NCLS; ++j) out[g * NCLS + j] = logits[j] - mx - lse;
}

// ---------------------------------------------------------------------------
// launch
// ---------------------------------------------------------------------------
extern "C" void kernel_launch(void* const* d_in, const int* in_sizes, int n_in,
                              void* d_out, int out_size, void* d_ws, size_t ws_size,
                              hipStream_t stream) {
  (void)n_in; (void)ws_size;
  const int N = in_sizes[0] / 2;
  const int E = in_sizes[1];
  const int G = out_size / NCLS;
  const int DENSE = in_sizes[13];       // fc1_b size
  const int HALF  = DENSE / 2;
  const int L     = in_sizes[7] / EMB;  // bs -> number of GCN layers

  const float* x     = (const float*)d_in[0];
  const float* ea    = (const float*)d_in[1];
  const int*   ei    = (const int*)d_in[2];
  const int*   batch = (const int*)d_in[3];
  const float* W0    = (const float*)d_in[4];
  const float* b0    = (const float*)d_in[5];
  const float* Ws    = (const float*)d_in[6];
  const float* bs    = (const float*)d_in[7];
  const float* g1    = (const float*)d_in[8];
  const float* beta1 = (const float*)d_in[9];
  const float* gs    = (const float*)d_in[10];
  const float* betas = (const float*)d_in[11];
  const float* fc1_w = (const float*)d_in[12];
  const float* fc1_b = (const float*)d_in[13];
  const float* g_fc1 = (const float*)d_in[14];
  const float* b_fc1 = (const float*)d_in[15];
  const float* fc2_w = (const float*)d_in[16];
  const float* fc2_b = (const float*)d_in[17];
  const float* g_fc2 = (const float*)d_in[18];
  const float* b_fc2 = (const float*)d_in[19];
  const float* fc3_w = (const float*)d_in[20];
  const float* fc3_b = (const float*)d_in[21];
  float* out = (float*)d_out;

  const int* src = ei;
  const int* dst = ei + E;

  size_t off = 0;
  auto alloc = [&](size_t bytes) -> void* {
    off = (off + 255) & ~(size_t)255;
    void* p = (char*)d_ws + off;
    off += bytes;
    return p;
  };
  float*  hW    = (float*)alloc((size_t)N * EMB * 4);
  float*  zbuf  = (float*)alloc((size_t)N * EMB * 4);
  __bf16* h_bf  = (__bf16*)alloc((size_t)N * EMB * 2);
  __bf16* Bp    = (__bf16*)alloc((size_t)L * EMB * EMB * 2); // packed weights
  float*  dinv  = (float*)alloc((size_t)N * 4);
  float*  enorm = (float*)alloc((size_t)E * 4);
  float*  s1    = (float*)alloc(256 * 4);
  float*  s2    = (float*)alloc(256 * 4);
  float*  scale = (float*)alloc(256 * 4);
  float*  shift = (float*)alloc(256 * 4);
  float*  gmp   = (float*)alloc((size_t)G * EMB * 4);
  float*  gap   = (float*)alloc((size_t)G * EMB * 4);
  float*  cnts  = (float*)alloc((size_t)G * 4);
  float*  rep   = (float*)alloc((size_t)G * 2 * EMB * 4);
  float*  wv1   = (float*)alloc((size_t)G * DENSE * 4);
  float*  wv2   = (float*)alloc((size_t)G * HALF * 4);

  const int B = 256;
  const long long NE = (long long)N * EMB;
  const long long GE = (long long)G * EMB;
  const long long scat = (long long)E * (EMB / 4);

  // degrees -> D^{-1/2}  (self-loop weight 1 baked in via fill)
  k_fill<<<cdiv(N, B), B, 0, stream>>>(dinv, 1.0f, N);
  k_deg<<<cdiv(E, B), B, 0, stream>>>(dst, ea, dinv, E);
  k_dinv<<<cdiv(N, B), B, 0, stream>>>(dinv, N);
  k_enorm<<<cdiv(E, B), B, 0, stream>>>(src, dst, ea, dinv, enorm, E);

  // graph sizes + rep accumulator + packed weights (all layers)
  k_fill<<<cdiv(G, B), B, 0, stream>>>(cnts, 0.f, G);
  k_counts<<<cdiv(N, B), B, 0, stream>>>(batch, cnts, N);
  k_fill<<<cdiv((long long)G * 2 * EMB, B), B, 0, stream>>>(rep, 0.f, (long long)G * 2 * EMB);
  for (int i = 0; i < L; ++i)
    k_pack_w<<<cdiv(EMB * EMB, B), B, 0, stream>>>(Ws + (size_t)i * EMB * EMB,
                                                   Bp + (size_t)i * EMB * EMB);

  for (int layer = 0; layer <= L; ++layer) {
    const float *bias, *gamma, *beta;
    if (layer == 0) {
      k_in_proj<<<cdiv(NE, B), B, 0, stream>>>(x, W0, hW, N);
      bias = b0; gamma = g1; beta = beta1;
    } else {
      int i = layer - 1;
      k_gemm_wmma<<<cdiv(N, 32), 256, 0, stream>>>(h_bf, Bp + (size_t)i * EMB * EMB, hW, N);
      bias = bs + i * EMB; gamma = gs + i * EMB; beta = betas + i * EMB;
    }
    k_fill<<<cdiv(NE, B), B, 0, stream>>>(zbuf, 0.f, NE);
    k_scatter<<<cdiv(scat, B), B, 0, stream>>>(hW, src, dst, enorm, zbuf, scat);
    k_fill<<<1, 256, 0, stream>>>(s1, 0.f, 256);
    k_fill<<<1, 256, 0, stream>>>(s2, 0.f, 256);
    k_bias_relu_stats<<<1024, EMB, 0, stream>>>(zbuf, hW, dinv, bias, s1, s2, N);
    k_bn_finalize<<<1, EMB, 0, stream>>>(s1, s2, gamma, beta, scale, shift, N, EMB);
    int do_pool = (layer >= 1);
    if (do_pool) {
      k_fill<<<cdiv(GE, B), B, 0, stream>>>(gap, 0.f, GE);
      k_fill<<<cdiv(GE, B), B, 0, stream>>>(gmp, -3.0e38f, GE);
    }
    k_bn_apply_pool<<<cdiv(NE, B), B, 0, stream>>>(zbuf, scale, shift, h_bf, batch, gmp, gap, N, do_pool);
    if (do_pool)
      k_pool_combine<<<cdiv(GE, B), B, 0, stream>>>(rep, gmp, gap, cnts, G);
  }

  // MLP head
  k_fc_relu<<<cdiv((long long)G * DENSE, B), B, 0, stream>>>(rep, fc1_w, fc1_b, wv1, G, 2 * EMB, DENSE);
  k_fill<<<1, 256, 0, stream>>>(s1, 0.f, 256);
  k_fill<<<1, 256, 0, stream>>>(s2, 0.f, 256);
  k_colstats<<<256, DENSE, 0, stream>>>(wv1, s1, s2, G, DENSE);
  k_bn_finalize<<<1, DENSE, 0, stream>>>(s1, s2, g_fc1, b_fc1, scale, shift, G, DENSE);
  k_bn_apply<<<cdiv((long long)G * DENSE, B), B, 0, stream>>>(wv1, scale, shift, (long long)G * DENSE, DENSE);

  k_fc_relu<<<cdiv((long long)G * HALF, B), B, 0, stream>>>(wv1, fc2_w, fc2_b, wv2, G, DENSE, HALF);
  k_fill<<<1, 256, 0, stream>>>(s1, 0.f, 256);
  k_fill<<<1, 256, 0, stream>>>(s2, 0.f, 256);
  k_colstats<<<256, HALF, 0, stream>>>(wv2, s1, s2, G, HALF);
  k_bn_finalize<<<1, HALF, 0, stream>>>(s1, s2, g_fc2, b_fc2, scale, shift, G, HALF);
  k_bn_apply<<<cdiv((long long)G * HALF, B), B, 0, stream>>>(wv2, scale, shift, (long long)G * HALF, HALF);

  k_head_out<<<cdiv(G, B), B, 0, stream>>>(wv2, fc3_w, fc3_b, out, G, HALF);
}
